// CGH_MMDecoderLSTM_44865228374203
// MI455X (gfx1250) — matrix operand order repaired and checked by reading, not compiled, not run on hardware
//
#include <hip/hip_runtime.h>
#include <hip/hip_bf16.h>

typedef _Float16 h8  __attribute__((ext_vector_type(8)));
typedef _Float16 v16h __attribute__((ext_vector_type(16)));
typedef float    v8f  __attribute__((ext_vector_type(8)));

#define SEQ 30
#define HD  64
#define ED  16
#define NM  3
#define DP  2
#define KCH 3          // K chunks of 32: h(64) + x(16) + [1.0, bias trick, pad] = 96
#define USTR 104       // padded row stride (halves) -> bank-conflict-free b128s
#define CONF_SEG (SEQ*DP + HD)   // 124 per-mode segment of conf input
#define KBIAS 80       // K slot carrying constant 1.0 (weights carry the bias)

// ---- fast activations: CDNA5 has a hardware v_tanh_f32 (TRANS32 op) ----
#if __has_builtin(__builtin_amdgcn_tanhf)
__device__ __forceinline__ float ftanh(float x) { return __builtin_amdgcn_tanhf(x); }
#else
__device__ __forceinline__ float ftanh(float x) {
    float t = __expf(2.0f * x);
    return fmaf(-2.0f, __builtin_amdgcn_rcpf(t + 1.0f), 1.0f);
}
#endif
__device__ __forceinline__ float fsigm(float x) {
    return fmaf(0.5f, ftanh(0.5f * x), 0.5f);
}

// A-fragment (16-bit A 16x32): lane holds halves [k0..k0+7] and [k0+16..k0+23]
__device__ __forceinline__ v16h ld_split16(const _Float16* p) {
    h8 a = *(const h8*)p;
    h8 b = *(const h8*)(p + 16);
    v16h r;
#pragma unroll
    for (int i = 0; i < 8; ++i) { r[i] = a[i]; r[i + 8] = b[i]; }
    return r;
}
// Contiguous loader: 16 consecutive halves p[0..15] (unused; B is loaded with ld_split16)
__device__ __forceinline__ v16h ld_cont16(const _Float16* p) {
    h8 a = *(const h8*)p;
    h8 b = *(const h8*)(p + 8);
    v16h r;
#pragma unroll
    for (int i = 0; i < 8; ++i) { r[i] = a[i]; r[i + 8] = b[i]; }
    return r;
}

__global__ __launch_bounds__(96)
void CGH_MMDecoderLSTM_kernel(const float* __restrict__ traj_rel,
                              const float* __restrict__ h0,
                              const float* __restrict__ c0,
                              const float* __restrict__ W_ih,
                              const float* __restrict__ W_hh,
                              const float* __restrict__ b_ih,
                              const float* __restrict__ b_hh,
                              const float* __restrict__ We,
                              const float* __restrict__ be,
                              const float* __restrict__ Wp,
                              const float* __restrict__ bp,
                              const float* __restrict__ Wc,
                              const float* __restrict__ bc,
                              float* __restrict__ out_pred,
                              float* __restrict__ out_conf)
{
    // Shared: combined weights [256 x 96 (pad 104)] f16 (cols: 0-63 W_hh, 64-79 W_ih,
    // 80 fused bias, 81-95 zero), per-mode state u=[h|x|1|0], conf/pred-head weights,
    // cross-wave conf partial scratch.
    __shared__ __align__(16) _Float16 Wl[4 * HD][USTR];        // 53.2 KB
    __shared__ __align__(16) _Float16 u[NM][16][USTR];         // 10.0 KB
    __shared__ float WcL[NM][NM * CONF_SEG];                   // 4.5 KB
    __shared__ float WpL[NM][DP][HD];                          // 1.5 KB
    __shared__ float confp[NM][16][NM];

    const int tid  = threadIdx.x;
    const int w    = tid >> 5;       // wave id == mode id
    const int lane = tid & 31;
    const int col  = lane & 15;      // batch-row (A) / column (B,D) index
    const int hv16 = lane >> 4;      // half-wave selector
    const int base = blockIdx.x * 16;
    const int m    = w;

    // ---------------- cooperative LDS init ----------------
    for (int idx = tid; idx < 4 * HD * 96; idx += 96) {
        int g = idx / 96, k = idx % 96;
        float v;
        if (k < HD)             v = W_hh[g * HD + k];
        else if (k < HD + ED)   v = W_ih[g * ED + (k - HD)];
        else if (k == KBIAS)    v = b_ih[g] + b_hh[g];   // fused bias column
        else                    v = 0.0f;
        Wl[g][k] = (_Float16)v;
    }
    for (int idx = tid; idx < NM * NM * CONF_SEG; idx += 96)
        WcL[idx / (NM * CONF_SEG)][idx % (NM * CONF_SEG)] = Wc[idx];
    for (int idx = tid; idx < NM * DP * HD; idx += 96)
        WpL[idx / (DP * HD)][(idx / HD) % DP][idx % HD] = Wp[idx];

    // ---------------- per-lane register constants ----------------
    float WeA[8], WeB[8], beR[8];                 // spatial embedding slice (e = hv16*8 + e8)
#pragma unroll
    for (int e8 = 0; e8 < 8; ++e8) {
        int e = hv16 * 8 + e8;
        WeA[e8] = We[e * DP + 0];
        WeB[e8] = We[e * DP + 1];
        beR[e8] = be[e];
    }
    float creg[4][8];                             // cell state, D-layout: row = v + 8*hv16
#pragma unroll
    for (int j = 0; j < 4; ++j)
#pragma unroll
        for (int v = 0; v < 8; ++v)
            creg[j][v] = c0[(base + v + 8 * hv16) * HD + j * 16 + col];

    // u init: h part + [1.0 | zeros] pad region; row = col
#pragma unroll 4
    for (int kk = 0; kk < 32; ++kk) {
        int k = hv16 * 32 + kk;
        u[m][col][k] = (_Float16)h0[(base + col) * HD + k];
    }
#pragma unroll
    for (int e8 = 0; e8 < 8; ++e8) {
        int k = HD + ED + hv16 * 8 + e8;
        u[m][col][k] = (_Float16)((k == KBIAS) ? 1.0f : 0.0f);
    }

    // x0 = leaky_relu(traj_rel @ We.T + be); lane (row=col, dim=hv16)
    const int d = hv16, row = col;
    const float bpR = bp[m * DP + d];
    {
        float rel  = traj_rel[(base + row) * DP + d];
        float relo = __shfl_xor(rel, 16, 32);
        float q0 = (d == 0) ? rel : relo;
        float q1 = (d == 0) ? relo : rel;
#pragma unroll
        for (int e8 = 0; e8 < 8; ++e8) {
            float x = fmaf(q0, WeA[e8], fmaf(q1, WeB[e8], beR[e8]));
            x = x > 0.0f ? x : 0.01f * x;
            u[m][row][HD + hv16 * 8 + e8] = (_Float16)x;
        }
    }
    float cp[NM] = {0.0f, 0.0f, 0.0f};            // on-the-fly confidence logit partials
    __syncthreads();

    // ---------------- 30-step recurrent loop (wave-private, no barriers) ----------------
#pragma unroll 1
    for (int s = 0; s < SEQ; ++s) {
        v16h A[KCH];
#pragma unroll
        for (int c = 0; c < KCH; ++c)
            A[c] = ld_split16(&u[m][row][c * 32 + hv16 * 8]);

#pragma unroll
        for (int j = 0; j < 4; ++j) {
            v8f acc[4];
#pragma unroll
            for (int g = 0; g < 4; ++g) {
                v8f a = {0.f, 0.f, 0.f, 0.f, 0.f, 0.f, 0.f, 0.f};
                const int n = g * 4 + j;          // gate tile: columns [n*16, n*16+16)
#pragma unroll
                for (int c = 0; c < KCH; ++c) {
                    v16h Bf = ld_split16(&Wl[n * 16 + col][c * 32 + hv16 * 8]);
                    a = __builtin_amdgcn_wmma_f32_16x16x32_f16(
                            false, A[c], false, Bf, (short)0, a, false, false);
                }
                acc[g] = a;
            }
#pragma unroll
            for (int v = 0; v < 8; ++v) {
                float iv = fsigm(acc[0][v]);      // bias already fused via K=80 column
                float fv = fsigm(acc[1][v]);
                float gv = ftanh(acc[2][v]);
                float ov = fsigm(acc[3][v]);
                float cn = fmaf(fv, creg[j][v], iv * gv);
                creg[j][v] = cn;
                float hn = ov * ftanh(cn);
                u[m][v + 8 * hv16][j * 16 + col] = (_Float16)hn;  // D-layout -> row-major
            }
        }

        // rel_pos = h_new @ Wp[m].T + bp[m]; lane handles (row, dim=d)
        float relv = bpR;
#pragma unroll
        for (int k8 = 0; k8 < 8; ++k8) {
            h8 hvv = *(const h8*)&u[m][row][k8 * 8];
#pragma unroll
            for (int i = 0; i < 8; ++i)
                relv = fmaf((float)hvv[i], WpL[m][d][k8 * 8 + i], relv);
        }
        out_pred[(((size_t)(base + row) * NM + m) * SEQ + s) * DP + d] = relv;
#pragma unroll
        for (int n = 0; n < NM; ++n)
            cp[n] = fmaf(WcL[n][m * CONF_SEG + s * DP + d], relv, cp[n]);

        // x_new = leaky_relu(rel_pos @ We.T + be)
        float ro = __shfl_xor(relv, 16, 32);
        float q0 = (d == 0) ? relv : ro;
        float q1 = (d == 0) ? ro : relv;
#pragma unroll
        for (int e8 = 0; e8 < 8; ++e8) {
            float x = fmaf(q0, WeA[e8], fmaf(q1, WeB[e8], beR[e8]));
            x = x > 0.0f ? x : 0.01f * x;
            u[m][row][HD + hv16 * 8 + e8] = (_Float16)x;
        }
    }

    // ---------------- confidence: final-h contribution + cross-mode softmax ----------------
#pragma unroll
    for (int k8 = 0; k8 < 4; ++k8) {
        h8 hvv = *(const h8*)&u[m][row][d * 32 + k8 * 8];
#pragma unroll
        for (int i = 0; i < 8; ++i) {
            int k = d * 32 + k8 * 8 + i;
            float hf = (float)hvv[i];
#pragma unroll
            for (int n = 0; n < NM; ++n)
                cp[n] = fmaf(WcL[n][m * CONF_SEG + SEQ * DP + k], hf, cp[n]);
        }
    }
#pragma unroll
    for (int n = 0; n < NM; ++n) cp[n] += __shfl_xor(cp[n], 16, 32);
    if (hv16 == 0) {
#pragma unroll
        for (int n = 0; n < NM; ++n) confp[m][row][n] = cp[n];
    }
    __syncthreads();

    if (tid < 16) {
        float lg[NM];
#pragma unroll
        for (int n = 0; n < NM; ++n)
            lg[n] = confp[0][tid][n] + confp[1][tid][n] + confp[2][tid][n] + bc[n];
        float mx = fmaxf(lg[0], fmaxf(lg[1], lg[2]));
        float e0 = __expf(lg[0] - mx), e1 = __expf(lg[1] - mx), e2 = __expf(lg[2] - mx);
        float inv = __builtin_amdgcn_rcpf(e0 + e1 + e2);
        out_conf[(size_t)(base + tid) * NM + 0] = e0 * inv;
        out_conf[(size_t)(base + tid) * NM + 1] = e1 * inv;
        out_conf[(size_t)(base + tid) * NM + 2] = e2 * inv;
    }
}

extern "C" void kernel_launch(void* const* d_in, const int* in_sizes, int n_in,
                              void* d_out, int out_size, void* d_ws, size_t ws_size,
                              hipStream_t stream) {
    (void)n_in; (void)d_ws; (void)ws_size; (void)out_size;
    // setup_inputs order: 0 traj_abs (unused), 1 traj_rel, 2 h0, 3 c0, 4 W_ih,
    // 5 W_hh, 6 b_ih, 7 b_hh, 8 We, 9 be, 10 Wp, 11 bp, 12 Wc, 13 bc
    const float* traj_rel = (const float*)d_in[1];
    const float* h0   = (const float*)d_in[2];
    const float* c0   = (const float*)d_in[3];
    const float* W_ih = (const float*)d_in[4];
    const float* W_hh = (const float*)d_in[5];
    const float* b_ih = (const float*)d_in[6];
    const float* b_hh = (const float*)d_in[7];
    const float* We   = (const float*)d_in[8];
    const float* be   = (const float*)d_in[9];
    const float* Wp   = (const float*)d_in[10];
    const float* bp   = (const float*)d_in[11];
    const float* Wc   = (const float*)d_in[12];
    const float* bc   = (const float*)d_in[13];

    const int Bn = in_sizes[2] / HD;              // 16384
    float* out_pred = (float*)d_out;
    float* out_conf = (float*)d_out + (size_t)Bn * NM * SEQ * DP;

    dim3 grid(Bn / 16), block(96);                // 3 waves/block: one per mode
    hipLaunchKernelGGL(CGH_MMDecoderLSTM_kernel, grid, block, 0, stream,
                       traj_rel, h0, c0, W_ih, W_hh, b_ih, b_hh,
                       We, be, Wp, bp, Wc, bc, out_pred, out_conf);
}
